// GATModel_39986145525985
// MI455X (gfx1250) — compile-verified, matrix operand order checked by reading, not run on hardware
//
#include <hip/hip_runtime.h>
#include <hip/hip_bf16.h>
#include <math.h>

// Problem constants (match reference)
#define NNODES 50000
#define NEDGES 500000
#define NB     64
#define INDIM  128
#define HIDD   32
#define NH     8
#define FEAT   256   // NH*HIDD
#define NCLS   10
#define SLOPE  0.2f

typedef __attribute__((ext_vector_type(16))) _Float16 v16h;
typedef __attribute__((ext_vector_type(8)))  _Float16 v8h;
typedef __attribute__((ext_vector_type(8)))  float    v8f;

// ---- order-preserving float -> uint key for atomic max on floats ----
static __device__ __forceinline__ unsigned fkey(float f) {
  unsigned u = __float_as_uint(f);
  return (u & 0x80000000u) ? ~u : (u | 0x80000000u);
}
static __device__ __forceinline__ float unfkey(unsigned k) {
  unsigned u = (k & 0x80000000u) ? (k & 0x7FFFFFFFu) : ~k;
  return __uint_as_float(u);
}

static __device__ __forceinline__ void atomic_fadd(float* p, float v) {
  __hip_atomic_fetch_add(p, v, __ATOMIC_RELAXED, __HIP_MEMORY_SCOPE_AGENT);
}

// ---------------------------------------------------------------- utilities
__global__ void k_zero_f32(float* __restrict__ p, int n) {
  int i = blockIdx.x * blockDim.x + threadIdx.x;
  if (i < n) p[i] = 0.0f;
}

__global__ void k_f32_to_f16(const float* __restrict__ x, _Float16* __restrict__ y, int n) {
  int i = blockIdx.x * blockDim.x + threadIdx.x;
  if (i < n) y[i] = (_Float16)x[i];
}

// W [K, 256] row-major  ->  Wt [256, K] row-major fp16
__global__ void k_w_transpose_f16(const float* __restrict__ W, _Float16* __restrict__ Wt, int K) {
  int i = blockIdx.x * blockDim.x + threadIdx.x;
  if (i >= FEAT * K) return;
  int n = i / K, k = i - n * K;
  Wt[i] = (_Float16)W[(size_t)k * FEAT + n];
}

// ---------------------------------------------------------------- WMMA GEMM
// C[M,256] = A[M,K] (fp16, row major) * B (fp16, given as Bt[256,K] row major).
// Each wave computes a 16(M) x 64(N) strip: one A fragment is reused across 4
// accumulators (4 WMMAs per A load -> 2.5 b128 loads per WMMA instead of 4).
// 16-bit A-fragment layout (ISA 7.12.2): lane<16 -> row=lane, K = {kb..kb+7, kb+16..kb+23}
//                                        lane>=16 -> row=lane-16, K = {kb+8..kb+15, kb+24..kb+31}
// B-fragment is the dual (lanes index column N, same K striping): Bt rows are B columns.
__global__ __launch_bounds__(128) void k_gemm_wmma(const _Float16* __restrict__ A,
                                                   const _Float16* __restrict__ Bt,
                                                   float* __restrict__ C,
                                                   int M, int K) {
  int wave = (blockIdx.x << 2) + (threadIdx.x >> 5);
  int tm = wave >> 2;     // M tile index (4 strips of 64 cols cover FEAT=256)
  int ts = wave & 3;      // strip index
  if (tm * 16 >= M) return;   // wave-uniform guard (EXEC stays all-ones)

  int lane = threadIdx.x & 31;
  int r    = lane & 15;
  int koff = (lane & 16) ? 8 : 0;

  const _Float16* arow = A  + (size_t)(tm * 16 + r) * K + koff;
  const _Float16* brow = Bt + (size_t)(ts * 64 + r) * K + koff;   // tile j adds j*16*K

  v8f acc0 = {}, acc1 = {}, acc2 = {}, acc3 = {};
  for (int kb = 0; kb < K; kb += 32) {
    v8h a0 = *(const v8h*)(arow + kb);        // 16B aligned: K%32==0, koff in {0,8}
    v8h a1 = *(const v8h*)(arow + kb + 16);
    v16h a = __builtin_shufflevector(a0, a1, 0,1,2,3,4,5,6,7,8,9,10,11,12,13,14,15);

#define BTILE(j, accv)                                                           \
    {                                                                            \
      const _Float16* bp = brow + (size_t)(j) * 16 * K + kb;                     \
      v8h b0 = *(const v8h*)(bp);                                                \
      v8h b1 = *(const v8h*)(bp + 16);                                           \
      v16h b = __builtin_shufflevector(b0, b1, 0,1,2,3,4,5,6,7,8,9,10,11,12,13,14,15); \
      accv = __builtin_amdgcn_wmma_f32_16x16x32_f16(false, a, false, b,          \
                                                    (short)0, accv, false, false); \
    }
    BTILE(0, acc0)
    BTILE(1, acc1)
    BTILE(2, acc2)
    BTILE(3, acc3)
#undef BTILE
  }

  // C/D layout: lane<16 -> col=lane, rows v+0 ; lane>=16 -> col=lane-16, rows v+8
  int ccol = lane & 15;
  int crow = (lane & 16) ? 8 : 0;
  float* cbase = C + (size_t)(tm * 16 + crow) * FEAT + ts * 64 + ccol;
#pragma unroll
  for (int v = 0; v < 8; ++v) {
    float* cp = cbase + (size_t)v * FEAT;
    cp[0]  = acc0[v];
    cp[16] = acc1[v];
    cp[32] = acc2[v];
    cp[48] = acc3[v];
  }
}

// ---------------------------------------------------------------- attention
// el/er: [N, H] = sum over HID of fs * al / ar
__global__ void k_elr(const float* __restrict__ fs, const float* __restrict__ al,
                      const float* __restrict__ ar, float* __restrict__ el,
                      float* __restrict__ er) {
  int i = blockIdx.x * blockDim.x + threadIdx.x;
  if (i >= NNODES * NH) return;
  int n = i >> 3, h = i & 7;
  const float* f = fs + (size_t)n * FEAT + h * HIDD;
  const float* a = al + h * HIDD;
  const float* b = ar + h * HIDD;
  float sl = 0.f, sr = 0.f;
#pragma unroll
  for (int d = 0; d < HIDD; ++d) { float v = f[d]; sl += v * a[d]; sr += v * b[d]; }
  el[i] = sl; er[i] = sr;
}

// per-edge logits + segment max over dst (via mapped u32 atomic max)
__global__ void k_edge_logit_max(const int* __restrict__ src, const int* __restrict__ dst,
                                 const float* __restrict__ el, const float* __restrict__ er,
                                 float* __restrict__ ebuf, unsigned* __restrict__ emax) {
  int i = blockIdx.x * blockDim.x + threadIdx.x;
  if (i >= NEDGES * NH) return;
  int e = i >> 3, h = i & 7;
  int s = src[e], d = dst[e];
  float v = el[s * NH + h] + er[d * NH + h];
  v = v > 0.f ? v : SLOPE * v;
  ebuf[i] = v;
  atomicMax(&emax[d * NH + h], fkey(v));
}

// exp(e - max) and segment sum of exps
__global__ void k_edge_expsum(const int* __restrict__ dst, float* __restrict__ ebuf,
                              const unsigned* __restrict__ emax, float* __restrict__ denom) {
  int i = blockIdx.x * blockDim.x + threadIdx.x;
  if (i >= NEDGES * NH) return;
  int e = i >> 3, h = i & 7;
  int d = dst[e];
  float m  = unfkey(emax[d * NH + h]);
  float ex = expf(ebuf[i] - m);
  ebuf[i]  = ex;
  atomic_fadd(&denom[d * NH + h], ex);
}

// alpha * fs[src] scattered into out[dst]; block = 256 threads = 8 heads x 32 dims
__global__ __launch_bounds__(256) void k_edge_aggr(const int* __restrict__ src,
                                                   const int* __restrict__ dst,
                                                   const float* __restrict__ ebuf,
                                                   const float* __restrict__ denom,
                                                   const float* __restrict__ fs,
                                                   float* __restrict__ out) {
  int e = blockIdx.x;
  int t = threadIdx.x;
  int h = t >> 5;
  int s = src[e], d = dst[e];
  float alpha = ebuf[e * NH + h] / denom[d * NH + h];
  float msg   = alpha * fs[(size_t)s * FEAT + t];
  atomic_fadd(&out[(size_t)d * FEAT + t], msg);
}

__global__ void k_bias_act(float* __restrict__ h, const float* __restrict__ b, int relu) {
  int i = blockIdx.x * blockDim.x + threadIdx.x;
  if (i >= NNODES * FEAT) return;
  float v = h[i] + b[i & (FEAT - 1)];
  if (relu) v = fmaxf(v, 0.f);
  h[i] = v;
}

// ---------------------------------------------------------------- pooling
__global__ void k_count(const int* __restrict__ n2g, int* __restrict__ counts) {
  int i = blockIdx.x * blockDim.x + threadIdx.x;
  if (i < NNODES) atomicAdd(&counts[n2g[i]], 1);
}

__global__ void k_pool_sum(const float* __restrict__ h, const int* __restrict__ n2g,
                           float* __restrict__ hg) {
  int i = blockIdx.x * blockDim.x + threadIdx.x;
  if (i >= NNODES * FEAT) return;
  int n = i >> 8;                 // FEAT == 256
  atomic_fadd(&hg[(size_t)n2g[n] * FEAT + (i & 255)], h[i]);
}

__global__ void k_classify(const float* __restrict__ hg, const int* __restrict__ counts,
                           const float* __restrict__ Wc, const float* __restrict__ bc,
                           float* __restrict__ out) {
  int i = blockIdx.x * blockDim.x + threadIdx.x;
  if (i >= NB * NCLS) return;
  int b = i / NCLS, c = i - b * NCLS;
  float cnt = (float)counts[b];
  if (cnt < 1.f) cnt = 1.f;
  float s = 0.f;
  for (int j = 0; j < FEAT; ++j) s += hg[b * FEAT + j] * Wc[j * NCLS + c];
  out[i] = s / cnt + bc[c];
}

// ---------------------------------------------------------------- host side
static inline size_t alignup(size_t x) { return (x + 255) & ~(size_t)255; }

static void run_gat_layer(const float* hin, int K, const float* W, const float* al,
                          const float* ar, const float* b, int relu,
                          _Float16* h16, _Float16* wt16, float* fs,
                          float* el, float* er, unsigned* emaxU, float* denom,
                          float* ebuf, float* hout,
                          const int* src, const int* dst, hipStream_t stream) {
  const int T = 256;
  int nIn = NNODES * K;
  k_f32_to_f16<<<(nIn + T - 1) / T, T, 0, stream>>>(hin, h16, nIn);
  k_w_transpose_f16<<<(FEAT * K + T - 1) / T, T, 0, stream>>>(W, wt16, K);
  // waves needed: (M/16) M-tiles * 4 strips; 4 waves per 128-thread block
  k_gemm_wmma<<<(NNODES / 16) * 4 / 4, 128, 0, stream>>>(h16, wt16, fs, NNODES, K);
  k_elr<<<(NNODES * NH + T - 1) / T, T, 0, stream>>>(fs, al, ar, el, er);
  k_zero_f32<<<(NNODES * NH + T - 1) / T, T, 0, stream>>>((float*)emaxU, NNODES * NH);
  k_zero_f32<<<(NNODES * NH + T - 1) / T, T, 0, stream>>>(denom, NNODES * NH);
  k_zero_f32<<<(NNODES * FEAT + T - 1) / T, T, 0, stream>>>(hout, NNODES * FEAT);
  k_edge_logit_max<<<(NEDGES * NH + T - 1) / T, T, 0, stream>>>(src, dst, el, er, ebuf, emaxU);
  k_edge_expsum<<<(NEDGES * NH + T - 1) / T, T, 0, stream>>>(dst, ebuf, emaxU, denom);
  k_edge_aggr<<<NEDGES, 256, 0, stream>>>(src, dst, ebuf, denom, fs, hout);
  k_bias_act<<<(NNODES * FEAT + T - 1) / T, T, 0, stream>>>(hout, b, relu);
}

extern "C" void kernel_launch(void* const* d_in, const int* in_sizes, int n_in,
                              void* d_out, int out_size, void* d_ws, size_t ws_size,
                              hipStream_t stream) {
  const float* feat = (const float*)d_in[0];
  const int*   src  = (const int*)d_in[1];
  const int*   dst  = (const int*)d_in[2];
  const int*   n2g  = (const int*)d_in[3];
  const float* W1   = (const float*)d_in[4];
  const float* al1  = (const float*)d_in[5];
  const float* ar1  = (const float*)d_in[6];
  const float* b1   = (const float*)d_in[7];
  const float* W2   = (const float*)d_in[8];
  const float* al2  = (const float*)d_in[9];
  const float* ar2  = (const float*)d_in[10];
  const float* b2   = (const float*)d_in[11];
  const float* Wc   = (const float*)d_in[12];
  const float* bc   = (const float*)d_in[13];
  float* out = (float*)d_out;

  char* p = (char*)d_ws;
  auto take = [&](size_t bytes) -> void* { void* r = (void*)p; p += alignup(bytes); return r; };
  float*     fs    = (float*)take((size_t)NNODES * FEAT * 4);
  float*     hbuf  = (float*)take((size_t)NNODES * FEAT * 4);
  _Float16*  h16   = (_Float16*)take((size_t)NNODES * FEAT * 2);
  _Float16*  wt16  = (_Float16*)take((size_t)FEAT * FEAT * 2);
  float*     el    = (float*)take((size_t)NNODES * NH * 4);
  float*     er    = (float*)take((size_t)NNODES * NH * 4);
  unsigned*  emaxU = (unsigned*)take((size_t)NNODES * NH * 4);
  float*     denom = (float*)take((size_t)NNODES * NH * 4);
  float*     ebuf  = (float*)take((size_t)NEDGES * NH * 4);
  float*     hg    = (float*)take((size_t)NB * FEAT * 4);
  int*       cnt   = (int*)take((size_t)NB * 4);

  // Layer 1: feat[N,128] -> hbuf[N,256] (ReLU)
  run_gat_layer(feat, INDIM, W1, al1, ar1, b1, /*relu=*/1,
                h16, wt16, fs, el, er, emaxU, denom, ebuf, hbuf, src, dst, stream);
  // Layer 2: hbuf -> hbuf (input consumed by the fp16 convert before hbuf is zeroed)
  run_gat_layer(hbuf, FEAT, W2, al2, ar2, b2, /*relu=*/0,
                h16, wt16, fs, el, er, emaxU, denom, ebuf, hbuf, src, dst, stream);

  // Mean pool per graph + classifier
  const int T = 256;
  k_zero_f32<<<(NB * FEAT + T - 1) / T, T, 0, stream>>>(hg, NB * FEAT);
  k_zero_f32<<<1, T, 0, stream>>>((float*)cnt, NB);
  k_count<<<(NNODES + T - 1) / T, T, 0, stream>>>(n2g, cnt);
  k_pool_sum<<<(NNODES * FEAT + T - 1) / T, T, 0, stream>>>(hbuf, n2g, hg);
  k_classify<<<(NB * NCLS + T - 1) / T, T, 0, stream>>>(hg, cnt, Wc, bc, out);
}
